// ExpertMLPs_48954037240375
// MI455X (gfx1250) — compile-verified, matrix operand order checked by reading, not run on hardware
//
#include <hip/hip_runtime.h>
#include <hip/hip_bf16.h>

// Problem constants (match reference)
#define T_TOK 8192
#define HID   1024
#define INTERM 4096
#define NEXP  8
#define TOPK  2
#define CAP   2048   // min(ceil(T*K*1.0/E), T)

typedef __attribute__((ext_vector_type(16))) _Float16 v16h;
typedef __attribute__((ext_vector_type(8)))  _Float16 v8h;
typedef __attribute__((ext_vector_type(4)))  _Float16 v4h;
typedef __attribute__((ext_vector_type(8)))  float    v8f;

#define BM 128
#define BN 128
#define BK 32
#define LDA 40   // LDS row stride in halfs: 80B rows (16B aligned for b128 pairs)

// --- CDNA5 async memory->LDS path (guarded: falls back to register staging) ---
#if defined(__gfx1250__) && __has_builtin(__builtin_amdgcn_global_load_async_to_lds_b128)
#define USE_ASYNC_LDS 1
#else
#define USE_ASYNC_LDS 0
#endif

#if __has_builtin(__builtin_amdgcn_s_wait_asynccnt)
#define WAIT_ASYNC() __builtin_amdgcn_s_wait_asynccnt(0)
#else
#define WAIT_ASYNC() asm volatile("s_wait_asynccnt 0" ::: "memory")
#endif

// 16-byte int vector in explicit address spaces (matches builtin signature:
// param 1 is 'int __attribute__((vector_size(16))) __device__ *', i.e. AS1)
typedef int b128v __attribute__((vector_size(16)));
typedef __attribute__((address_space(1))) b128v* gas_b128;
typedef __attribute__((address_space(3))) b128v* las_b128;

// ---------------------------------------------------------------------------
// Kernel 1: routing. One block, 256 threads = 8 waves; wave w == expert w.
// ---------------------------------------------------------------------------
__global__ __launch_bounds__(256) void moe_route_kernel(
    const int*   __restrict__ eidx,    // (T, 2)
    const float* __restrict__ aff,     // (T, 8)
    int*            __restrict__ assign,  // (E*CAP) slot -> token (default 0)
    unsigned short* __restrict__ pos16,   // (T, 8) kept 1-based position, 0 = dropped
    int*            __restrict__ perm,    // (T, 2) slot index (0-based)
    float*          __restrict__ affk)    // (T, 2) normalized affinity
{
    const int tid  = threadIdx.x;
    const int lane = tid & 31;
    const int wid  = tid >> 5;

    {
        const int e = wid;  // exactly 8 waves
        for (int i = lane; i < CAP; i += 32) assign[e * CAP + i] = 0;

        int base = 0;
        for (int t0 = 0; t0 < T_TOK; t0 += 32) {
            const int t  = t0 + lane;
            const int i0 = eidx[2 * t];
            const int i1 = eidx[2 * t + 1];
            const int m  = (i0 == e) + (i1 == e);   // multiplicity (0,1,2)
            int v = m;
            #pragma unroll
            for (int d = 1; d < 32; d <<= 1) {
                int n = __shfl_up(v, d, 32);
                if (lane >= d) v += n;
            }
            const int pos  = base + v;              // inclusive cumsum (1-based)
            const int kept = (m > 0) && (pos <= CAP);
            if (kept) {
                assign[e * CAP + pos - 1] = t;
                if (m == 2) assign[e * CAP + pos - 2] = 0;  // gap slot -> token 0
            }
            pos16[t * NEXP + e] = kept ? (unsigned short)pos : (unsigned short)0;
            base += __shfl(v, 31, 32);
        }
    }
    __threadfence();
    __syncthreads();

    for (int t = tid; t < T_TOK; t += 256) {
        unsigned short p[NEXP];
        float denom = 0.f;
        #pragma unroll
        for (int e = 0; e < NEXP; ++e) {
            p[e] = pos16[t * NEXP + e];
            if (p[e]) denom += fabsf(aff[t * NEXP + e]);
        }
        denom = fmaxf(denom, 1e-12f);
        #pragma unroll
        for (int k = 0; k < TOPK; ++k) {
            const int e  = eidx[2 * t + k];
            const int pp = p[e];
            perm[2 * t + k] = pp ? (e * CAP + pp - 1) : 0;
            affk[2 * t + k] = pp ? (aff[t * NEXP + e] / denom) : 0.f;
        }
    }
}

// ---------------------------------------------------------------------------
// Kernel 2: fused gate/up GEMM + SiLU-GLU, double-buffered LDS pipeline.
// inter[e,c,i] = silu(x W_g) * (x W_u)
// ---------------------------------------------------------------------------
__global__ __launch_bounds__(256) void moe_gemm1_kernel(
    const float* __restrict__ hidden,   // (T, H) fp32
    const float* __restrict__ gate_w,   // (E, H, I) fp32
    const float* __restrict__ up_w,     // (E, H, I) fp32
    const int*   __restrict__ assign,   // (E*CAP)
    _Float16*    __restrict__ inter)    // (E, CAP, I) f16
{
    constexpr int TILE_H = BM * LDA;    // 5120 halfs per tile
    constexpr int BUF_H  = 3 * TILE_H;  // A + G + U
    __shared__ __align__(32) _Float16 smem[2 * BUF_H];  // 61,440 B

    const int e   = blockIdx.z;
    const int m0  = blockIdx.y * BM;
    const int n0  = blockIdx.x * BN;
    const int tid = threadIdx.x;
    const int lane = tid & 31;
    const int wid  = tid >> 5;
    const int wm = (wid >> 2) * 64;
    const int wn = (wid & 3) * 32;
    const int fr = lane & 15;
    const int kh = (lane >> 4) * 16;

    const int arow = tid >> 1;
    const int acol = (tid & 1) << 4;
    const int tok  = assign[e * CAP + m0 + arow];
    const float* aptr = hidden + (size_t)tok * HID + acol;

    const int kk = (tid >> 5) << 2;
    const int nn = (tid & 31) << 2;
    const float* gptr = gate_w + ((size_t)e * HID + kk) * INTERM + n0 + nn;
    const float* uptr = up_w   + ((size_t)e * HID + kk) * INTERM + n0 + nn;

    float4 ra[4], rg[4], ru[4];

    auto load_tile = [&](int k0) {
        const float* pa = aptr + k0;
        ra[0] = *(const float4*)(pa);
        ra[1] = *(const float4*)(pa + 4);
        ra[2] = *(const float4*)(pa + 8);
        ra[3] = *(const float4*)(pa + 12);
        const float* pg = gptr + (size_t)k0 * INTERM;
        rg[0] = *(const float4*)(pg);
        rg[1] = *(const float4*)(pg + INTERM);
        rg[2] = *(const float4*)(pg + 2 * INTERM);
        rg[3] = *(const float4*)(pg + 3 * INTERM);
        const float* pu = uptr + (size_t)k0 * INTERM;
        ru[0] = *(const float4*)(pu);
        ru[1] = *(const float4*)(pu + INTERM);
        ru[2] = *(const float4*)(pu + 2 * INTERM);
        ru[3] = *(const float4*)(pu + 3 * INTERM);
    };

    auto store_tile = [&](int b) {
        _Float16* base = smem + b * BUF_H;
        _Float16* dA = base + arow * LDA + acol;
        *(v4h*)(dA + 0)  = {(_Float16)ra[0].x, (_Float16)ra[0].y, (_Float16)ra[0].z, (_Float16)ra[0].w};
        *(v4h*)(dA + 4)  = {(_Float16)ra[1].x, (_Float16)ra[1].y, (_Float16)ra[1].z, (_Float16)ra[1].w};
        *(v4h*)(dA + 8)  = {(_Float16)ra[2].x, (_Float16)ra[2].y, (_Float16)ra[2].z, (_Float16)ra[2].w};
        *(v4h*)(dA + 12) = {(_Float16)ra[3].x, (_Float16)ra[3].y, (_Float16)ra[3].z, (_Float16)ra[3].w};
        _Float16* dG = base + TILE_H + nn * LDA + kk;   // transposed [n][k]
        *(v4h*)(dG + 0 * LDA) = {(_Float16)rg[0].x, (_Float16)rg[1].x, (_Float16)rg[2].x, (_Float16)rg[3].x};
        *(v4h*)(dG + 1 * LDA) = {(_Float16)rg[0].y, (_Float16)rg[1].y, (_Float16)rg[2].y, (_Float16)rg[3].y};
        *(v4h*)(dG + 2 * LDA) = {(_Float16)rg[0].z, (_Float16)rg[1].z, (_Float16)rg[2].z, (_Float16)rg[3].z};
        *(v4h*)(dG + 3 * LDA) = {(_Float16)rg[0].w, (_Float16)rg[1].w, (_Float16)rg[2].w, (_Float16)rg[3].w};
        _Float16* dU = base + 2 * TILE_H + nn * LDA + kk;
        *(v4h*)(dU + 0 * LDA) = {(_Float16)ru[0].x, (_Float16)ru[1].x, (_Float16)ru[2].x, (_Float16)ru[3].x};
        *(v4h*)(dU + 1 * LDA) = {(_Float16)ru[0].y, (_Float16)ru[1].y, (_Float16)ru[2].y, (_Float16)ru[3].y};
        *(v4h*)(dU + 2 * LDA) = {(_Float16)ru[0].z, (_Float16)ru[1].z, (_Float16)ru[2].z, (_Float16)ru[3].z};
        *(v4h*)(dU + 3 * LDA) = {(_Float16)ru[0].w, (_Float16)ru[1].w, (_Float16)ru[2].w, (_Float16)ru[3].w};
    };

    v8f accG[4][2], accU[4][2];
    #pragma unroll
    for (int i = 0; i < 4; ++i)
        #pragma unroll
        for (int j = 0; j < 2; ++j) { accG[i][j] = {}; accU[i][j] = {}; }

    load_tile(0);
    store_tile(0);
    __syncthreads();

    constexpr int NIT = HID / BK;   // 32
    for (int it = 0; it < NIT; ++it) {
        const int cur = it & 1;
        if (it + 1 < NIT) load_tile((it + 1) * BK);

        const _Float16* base = smem + cur * BUF_H;
        v16h av[4], bg[2], bu[2];
        #pragma unroll
        for (int fm = 0; fm < 4; ++fm)
            av[fm] = *(const v16h*)(base + (wm + fm * 16 + fr) * LDA + kh);
        #pragma unroll
        for (int fn = 0; fn < 2; ++fn) {
            bg[fn] = *(const v16h*)(base + TILE_H     + (wn + fn * 16 + fr) * LDA + kh);
            bu[fn] = *(const v16h*)(base + 2 * TILE_H + (wn + fn * 16 + fr) * LDA + kh);
        }
        #pragma unroll
        for (int fm = 0; fm < 4; ++fm)
            #pragma unroll
            for (int fn = 0; fn < 2; ++fn) {
                accG[fm][fn] = __builtin_amdgcn_wmma_f32_16x16x32_f16(
                    false, av[fm], false, bg[fn], (short)0, accG[fm][fn], false, false);
                accU[fm][fn] = __builtin_amdgcn_wmma_f32_16x16x32_f16(
                    false, av[fm], false, bu[fn], (short)0, accU[fm][fn], false, false);
            }
        if (it + 1 < NIT) store_tile(cur ^ 1);
        __syncthreads();
    }

    // ---- epilogue: silu(gate)*up -> LDS bounce -> coalesced f16 stores ----
    {
        constexpr int LDO = 136;      // padded f16 row stride (272B)
        _Float16* sOut = smem;        // 128*136 halfs = 34,816B <= 61,440B
        #pragma unroll
        for (int fm = 0; fm < 4; ++fm)
            #pragma unroll
            for (int fn = 0; fn < 2; ++fn)
                #pragma unroll
                for (int r = 0; r < 8; ++r) {
                    const int m  = wm + fm * 16 + ((lane < 16) ? r : r + 8);
                    const int nl = wn + fn * 16 + fr;
                    const float g = accG[fm][fn][r];
                    const float u = accU[fm][fn][r];
                    const float s = g / (1.0f + __expf(-g));
                    sOut[m * LDO + nl] = (_Float16)(s * u);
                }
        __syncthreads();
        const int orow = tid >> 1;
        const int ocol = (tid & 1) << 6;   // 0 | 64
        const _Float16* src = sOut + orow * LDO + ocol;
        _Float16* dst = inter + ((size_t)e * CAP + m0 + orow) * INTERM + n0 + ocol;
        #pragma unroll
        for (int j = 0; j < 8; ++j)
            *(v8h*)(dst + 8 * j) = *(const v8h*)(src + 8 * j);
    }
}

// ---------------------------------------------------------------------------
// Kernel 3: down GEMM, double-buffered; A tile staged via async memory->LDS.
// ---------------------------------------------------------------------------
__global__ __launch_bounds__(256) void moe_gemm2_kernel(
    const _Float16* __restrict__ inter,   // (E, CAP, I) f16
    const float*    __restrict__ down_w,  // (E, I, H) fp32
    float*          __restrict__ pout)    // (E*CAP, H) fp32
{
    constexpr int TILE_H = BM * LDA;      // 5120 halfs
    constexpr int BUF_H  = 2 * TILE_H;    // A + B
    __shared__ __align__(32) _Float16 smem[2 * BUF_H];  // 40,960 B

    const int e   = blockIdx.z;
    const int m0  = blockIdx.y * BM;
    const int n0  = blockIdx.x * BN;
    const int tid = threadIdx.x;
    const int lane = tid & 31;
    const int wid  = tid >> 5;
    const int wm = (wid >> 2) * 64;
    const int wn = (wid & 3) * 32;
    const int fr = lane & 15;
    const int kh = (lane >> 4) * 16;

    const int arow = tid >> 1;
    const int acol = (tid & 1) << 4;
    const _Float16* aptr = inter + ((size_t)e * CAP + m0 + arow) * INTERM + acol;

    const int kk = (tid >> 5) << 2;
    const int nn = (tid & 31) << 2;
    const float* bptr = down_w + ((size_t)e * INTERM + kk) * HID + n0 + nn;

    float4 rb[4];
    auto load_B = [&](int k0) {
        const float* p = bptr + (size_t)k0 * HID;
        rb[0] = *(const float4*)(p);
        rb[1] = *(const float4*)(p + HID);
        rb[2] = *(const float4*)(p + 2 * HID);
        rb[3] = *(const float4*)(p + 3 * HID);
    };
    auto store_B = [&](int b) {
        _Float16* dB = smem + b * BUF_H + TILE_H + nn * LDA + kk;
        *(v4h*)(dB + 0 * LDA) = {(_Float16)rb[0].x, (_Float16)rb[1].x, (_Float16)rb[2].x, (_Float16)rb[3].x};
        *(v4h*)(dB + 1 * LDA) = {(_Float16)rb[0].y, (_Float16)rb[1].y, (_Float16)rb[2].y, (_Float16)rb[3].y};
        *(v4h*)(dB + 2 * LDA) = {(_Float16)rb[0].z, (_Float16)rb[1].z, (_Float16)rb[2].z, (_Float16)rb[3].z};
        *(v4h*)(dB + 3 * LDA) = {(_Float16)rb[0].w, (_Float16)rb[1].w, (_Float16)rb[2].w, (_Float16)rb[3].w};
    };

#if USE_ASYNC_LDS
    auto copy_A_async = [&](int k0, int b) {
        const _Float16* src = aptr + k0;
        _Float16* dst = smem + b * BUF_H + arow * LDA + acol;
        __builtin_amdgcn_global_load_async_to_lds_b128((gas_b128)(src),     (las_b128)(dst),     0, 0);
        __builtin_amdgcn_global_load_async_to_lds_b128((gas_b128)(src + 8), (las_b128)(dst + 8), 0, 0);
    };
#else
    v8h ra0, ra1;
    auto load_A = [&](int k0) {
        ra0 = *(const v8h*)(aptr + k0);
        ra1 = *(const v8h*)(aptr + k0 + 8);
    };
    auto store_A = [&](int b) {
        _Float16* dA = smem + b * BUF_H + arow * LDA + acol;
        *(v8h*)(dA)     = ra0;
        *(v8h*)(dA + 8) = ra1;
    };
#endif

    v8f acc[4][2];
    #pragma unroll
    for (int i = 0; i < 4; ++i)
        #pragma unroll
        for (int j = 0; j < 2; ++j) acc[i][j] = {};

    // prologue: stage tile 0
    load_B(0);
    store_B(0);
#if USE_ASYNC_LDS
    copy_A_async(0, 0);
    WAIT_ASYNC();
#else
    load_A(0);
    store_A(0);
#endif
    __syncthreads();

    constexpr int NIT = INTERM / BK;   // 128
    for (int it = 0; it < NIT; ++it) {
        const int cur = it & 1;
        const bool has_next = (it + 1) < NIT;
        if (has_next) {
            load_B((it + 1) * BK);
#if USE_ASYNC_LDS
            copy_A_async((it + 1) * BK, cur ^ 1);
#else
            load_A((it + 1) * BK);
#endif
        }

        const _Float16* base = smem + cur * BUF_H;
        v16h av[4], bv[2];
        #pragma unroll
        for (int fm = 0; fm < 4; ++fm)
            av[fm] = *(const v16h*)(base + (wm + fm * 16 + fr) * LDA + kh);
        #pragma unroll
        for (int fn = 0; fn < 2; ++fn)
            bv[fn] = *(const v16h*)(base + TILE_H + (wn + fn * 16 + fr) * LDA + kh);
        #pragma unroll
        for (int fm = 0; fm < 4; ++fm)
            #pragma unroll
            for (int fn = 0; fn < 2; ++fn)
                acc[fm][fn] = __builtin_amdgcn_wmma_f32_16x16x32_f16(
                    false, av[fm], false, bv[fn], (short)0, acc[fm][fn], false, false);

        if (has_next) {
            store_B(cur ^ 1);
#if !USE_ASYNC_LDS
            store_A(cur ^ 1);
#endif
        }
#if USE_ASYNC_LDS
        WAIT_ASYNC();
#endif
        __syncthreads();
    }

    #pragma unroll
    for (int fm = 0; fm < 4; ++fm)
        #pragma unroll
        for (int fn = 0; fn < 2; ++fn)
            #pragma unroll
            for (int r = 0; r < 8; ++r) {
                const int m = wm + fm * 16 + ((lane < 16) ? r : r + 8);
                const int n = n0 + wn + fn * 16 + fr;
                pout[((size_t)e * CAP + m0 + m) * HID + n] = acc[fm][fn][r];
            }
}

// ---------------------------------------------------------------------------
// Kernel 4: combine. out[t] = sum_k aff_k * pout[perm_k]
// ---------------------------------------------------------------------------
__global__ __launch_bounds__(256) void moe_combine_kernel(
    const float* __restrict__ pout,
    const int*   __restrict__ perm,
    const float* __restrict__ affk,
    float*       __restrict__ out)
{
    const int t = blockIdx.x;
    const int x = threadIdx.x;             // 256 threads * float4 == 1024 floats
    const float a0 = affk[2 * t];
    const float a1 = affk[2 * t + 1];
    const float4* r0 = (const float4*)(pout + (size_t)perm[2 * t] * HID);
    const float4* r1 = (const float4*)(pout + (size_t)perm[2 * t + 1] * HID);
    float4 v0 = r0[x];
    float4 v1 = r1[x];
    float4 o;
    o.x = a0 * v0.x + a1 * v1.x;
    o.y = a0 * v0.y + a1 * v1.y;
    o.z = a0 * v0.z + a1 * v1.z;
    o.w = a0 * v0.w + a1 * v1.w;
    ((float4*)(out + (size_t)t * HID))[x] = o;
}

// ---------------------------------------------------------------------------
extern "C" void kernel_launch(void* const* d_in, const int* in_sizes, int n_in,
                              void* d_out, int out_size, void* d_ws, size_t ws_size,
                              hipStream_t stream) {
    const float* hidden = (const float*)d_in[0];
    const float* aff    = (const float*)d_in[1];
    const float* gate_w = (const float*)d_in[2];
    const float* up_w   = (const float*)d_in[3];
    const float* down_w = (const float*)d_in[4];
    const int*   eidx   = (const int*)d_in[5];
    float* out = (float*)d_out;

    // workspace layout
    char* ws = (char*)d_ws;
    int*            assign = (int*)(ws + 0);                         // 64 KB
    int*            perm   = (int*)(ws + (64 << 10));                // 64 KB
    float*          affk   = (float*)(ws + (128 << 10));             // 64 KB
    unsigned short* pos16  = (unsigned short*)(ws + (192 << 10));    // 128 KB
    _Float16*       inter  = (_Float16*)(ws + (320 << 10));          // 128 MB
    float*          pout   = (float*)(ws + (320 << 10)
                                      + (size_t)NEXP * CAP * INTERM * sizeof(_Float16)); // 64 MB

    moe_route_kernel<<<1, 256, 0, stream>>>(eidx, aff, assign, pos16, perm, affk);

    dim3 g1(INTERM / BN, CAP / BM, NEXP);   // (32, 16, 8)
    moe_gemm1_kernel<<<g1, 256, 0, stream>>>(hidden, gate_w, up_w, assign, inter);

    dim3 g2(HID / BN, CAP / BM, NEXP);      // (8, 16, 8)
    moe_gemm2_kernel<<<g2, 256, 0, stream>>>(inter, down_w, pout);

    moe_combine_kernel<<<T_TOK, 256, 0, stream>>>(pout, perm, affk, out);
}
